// Decoder_56384330662539
// MI455X (gfx1250) — matrix operand order repaired and checked by reading, not compiled, not run on hardware
//
#include <hip/hip_runtime.h>
#include <stdint.h>
#include <stddef.h>

#define SOS_IDX 1
#define B_ 64
#define S_ 64
#define H_ 512
#define E_ 512
#define V_ 8000
#define VH_ (V_ + H_)   // 8512

typedef __attribute__((ext_vector_type(16))) __bf16    v16bf_t;
typedef __attribute__((ext_vector_type(16))) uint16_t  v16u_t;
typedef __attribute__((ext_vector_type(8)))  float     v8f_t;

// ---------------- bf16 helpers ----------------

__device__ __forceinline__ uint16_t f2bf(float f) {
  uint32_t x = __builtin_bit_cast(uint32_t, f);
  x += 0x7FFFu + ((x >> 16) & 1u);        // round-to-nearest-even
  return (uint16_t)(x >> 16);
}

// A fragment: 16x32 bf16, row-major (M,K) source, base already at row m0.
// Layout: lanes 0-15 => M=lane, K = 0..15; lanes 16-31 => same M, K = 16..31.
__device__ __forceinline__ v16bf_t load_a16(const uint16_t* base, size_t ld, int k0, int lane) {
  const uint16_t* p = base + (size_t)(lane & 15) * ld + (size_t)k0 + (size_t)((lane >> 4) << 4);
  v16u_t u;
#pragma unroll
  for (int i = 0; i < 8; ++i) u[i] = p[i];
#pragma unroll
  for (int i = 0; i < 8; ++i) u[8 + i] = p[8 + i];
  return __builtin_bit_cast(v16bf_t, u);
}

// B fragment: 32x16 bf16; source stored as B^T row-major (N,K), base at row n0.
// Layout: lanes 0-15 => N=lane, K = 0..15; lanes 16-31 => same N, K = 16..31.
__device__ __forceinline__ v16bf_t load_b16(const uint16_t* baseNT, size_t ld, int k0, int lane) {
  const uint16_t* p = baseNT + (size_t)(lane & 15) * ld + (size_t)k0 + (size_t)((lane >> 4) << 4);
  v16u_t u;
#pragma unroll
  for (int i = 0; i < 16; ++i) u[i] = p[i];
  return __builtin_bit_cast(v16bf_t, u);
}

// B fragment converting from fp32 source (B^T row-major (N,K)).
__device__ __forceinline__ v16bf_t load_b16_f32(const float* baseNT, size_t ld, int k0, int lane) {
  const float* p = baseNT + (size_t)(lane & 15) * ld + (size_t)k0 + (size_t)((lane >> 4) << 4);
  v16u_t u;
#pragma unroll
  for (int i = 0; i < 16; ++i) u[i] = f2bf(p[i]);
  return __builtin_bit_cast(v16bf_t, u);
}

__device__ __forceinline__ v8f_t wmma_bf16(v16bf_t a, v16bf_t b, v8f_t c) {
  return __builtin_amdgcn_wmma_f32_16x16x32_bf16(false, a, false, b, (short)0, c, false, false);
}

// ---------------- one-time prep kernels ----------------

__global__ void k_convert(uint16_t* __restrict__ dst, const float* __restrict__ src, int n) {
  int i = blockIdx.x * blockDim.x + threadIdx.x;
  if (i < n) dst[i] = f2bf(src[i]);
}

// src (rows, cols) fp32 -> dst (cols, rows) bf16
__global__ void k_transpose_convert(uint16_t* __restrict__ dst, const float* __restrict__ src,
                                    int rows, int cols) {
  int i = blockIdx.x * blockDim.x + threadIdx.x;
  if (i < rows * cols) {
    int r = i / cols, c = i % cols;
    dst[(size_t)c * rows + r] = f2bf(src[i]);
  }
}

// dst[r,c] = bf16(src[r*src_ld + src_off + c]), dst (rows, cols) contiguous
__global__ void k_slice_convert(uint16_t* __restrict__ dst, const float* __restrict__ src,
                                int rows, int cols, int src_ld, int src_off) {
  int i = blockIdx.x * blockDim.x + threadIdx.x;
  if (i < rows * cols) {
    int r = i / cols, c = i % cols;
    dst[i] = f2bf(src[(size_t)r * src_ld + src_off + c]);
  }
}

// x_all[b,s,e] = bf16(emb[token(b,s), e]); token(b,0)=SOS, else target[b,s]
__global__ void k_embed(uint16_t* __restrict__ x_all, const int* __restrict__ target,
                        const float* __restrict__ emb) {
  int i = blockIdx.x * blockDim.x + threadIdx.x;
  if (i >= B_ * S_ * E_) return;
  int e = i % E_;
  int bs = i / E_;
  int s = bs % S_;
  int b = bs / S_;
  int tok = (s == 0) ? SOS_IDX : target[b * S_ + s];
  x_all[i] = f2bf(emb[(size_t)tok * E_ + e]);
}

// h0 = encoder_hiddens[:, S-1, :]
__global__ void k_h0(float* __restrict__ hf, uint16_t* __restrict__ hb,
                     const float* __restrict__ enc) {
  int i = blockIdx.x * blockDim.x + threadIdx.x;
  if (i >= B_ * H_) return;
  int h = i % H_;
  int b = i / H_;
  float v = enc[((size_t)b * S_ + (S_ - 1)) * H_ + h];
  hf[i] = v;
  hb[i] = f2bf(v);
}

// bfused[j] = b_c[j] + sum_k W_c[j,k] * b_ho[k]
__global__ void k_bias_fuse(float* __restrict__ bf, const float* __restrict__ W_c,
                            const float* __restrict__ b_ho, const float* __restrict__ b_c) {
  int j = blockIdx.x * blockDim.x + threadIdx.x;
  if (j >= V_) return;
  const float* row = W_c + (size_t)j * VH_;
  float acc = b_c[j];
  for (int k = 0; k < V_; ++k) acc += row[k] * b_ho[k];
  bf[j] = acc;
}

// Fusion GEMM (transposed form): compute Wf^T (H x V) = W_hoT @ W_c[:, :V]^T,
// store transposed into Wf (V x H) bf16.
//   M = H (A = W_hoT bf16, (H x V) row-major, contiguous K)
//   N = V (B^T = W_c[:, :V] fp32, ld = VH_, converted per fragment)
//   K = V
// One wave computes 4 M-tiles (64x16 of Wf^T) sharing each converted B fragment.
__global__ __launch_bounds__(256)
void k_wf_gemm(uint16_t* __restrict__ Wf, const float* __restrict__ W_c,
               const uint16_t* __restrict__ W_hoT) {
  int lane = threadIdx.x & 31;
  int wid  = threadIdx.x >> 5;
  int jid  = blockIdx.x * 8 + wid;        // 8 m-groups * 500 n-tiles = 4000 jobs
  int nt = jid % 500;                     // V tile
  int mg = jid / 500;                     // group of 4 H tiles
  const uint16_t* A0 = W_hoT + (size_t)(mg * 64 +  0) * V_;
  const uint16_t* A1 = W_hoT + (size_t)(mg * 64 + 16) * V_;
  const uint16_t* A2 = W_hoT + (size_t)(mg * 64 + 32) * V_;
  const uint16_t* A3 = W_hoT + (size_t)(mg * 64 + 48) * V_;
  const float*    Bb = W_c + (size_t)(nt * 16) * VH_;
  v8f_t acc0 = {}, acc1 = {}, acc2 = {}, acc3 = {};
  for (int k0 = 0; k0 < V_; k0 += 32) {
    v16bf_t b = load_b16_f32(Bb, VH_, k0, lane);
    acc0 = wmma_bf16(load_a16(A0, V_, k0, lane), b, acc0);
    acc1 = wmma_bf16(load_a16(A1, V_, k0, lane), b, acc1);
    acc2 = wmma_bf16(load_a16(A2, V_, k0, lane), b, acc2);
    acc3 = wmma_bf16(load_a16(A3, V_, k0, lane), b, acc3);
  }
  // D tile (m = h index, n = j index) written transposed into Wf[j*H + h]:
  // per lane 8 consecutive bf16 (16B aligned) per accumulator.
  int j  = nt * 16 + (lane & 15);
  uint16_t* row = Wf + (size_t)j * H_ + (size_t)mg * 64 + (size_t)((lane >> 4) << 3);
  const v8f_t accs[4] = {acc0, acc1, acc2, acc3};
#pragma unroll
  for (int i = 0; i < 4; ++i)
#pragma unroll
    for (int r = 0; r < 8; ++r)
      row[i * 16 + r] = f2bf(accs[i][r]);
}

// ---------------- per-step kernels ----------------

// h_new = tanh(x_t @ W_ih^T + h_prev @ W_hh^T + b_cell)   (64 x 512)
__global__ __launch_bounds__(128)
void k_step_h(float* __restrict__ h_new_f, uint16_t* __restrict__ h_new_b,
              const uint16_t* __restrict__ x_all, int t,
              const uint16_t* __restrict__ h_prev_b,
              const uint16_t* __restrict__ W_ih_b, const uint16_t* __restrict__ W_hh_b,
              const float* __restrict__ b_cell) {
  int lane = threadIdx.x & 31;
  int wid  = threadIdx.x >> 5;
  int tile = blockIdx.x * 4 + wid;        // 4*32 = 128 tiles
  int nt = tile & 31;                     // H tiles
  int mt = tile >> 5;                     // B tiles
  const uint16_t* Ax = x_all    + (size_t)(mt * 16) * ((size_t)S_ * E_) + (size_t)t * E_;
  const uint16_t* Bx = W_ih_b   + (size_t)(nt * 16) * E_;
  const uint16_t* Ah = h_prev_b + (size_t)(mt * 16) * H_;
  const uint16_t* Bh = W_hh_b   + (size_t)(nt * 16) * H_;
  v8f_t acc = {};
  for (int k0 = 0; k0 < E_; k0 += 32)
    acc = wmma_bf16(load_a16(Ax, (size_t)S_ * E_, k0, lane), load_b16(Bx, E_, k0, lane), acc);
  for (int k0 = 0; k0 < H_; k0 += 32)
    acc = wmma_bf16(load_a16(Ah, H_, k0, lane), load_b16(Bh, H_, k0, lane), acc);
  int n  = nt * 16 + (lane & 15);
  int mo = mt * 16 + ((lane >> 4) << 3);
  float bn = b_cell[n];
#pragma unroll
  for (int r = 0; r < 8; ++r) {
    float v = tanhf(acc[r] + bn);
    size_t idx = (size_t)(mo + r) * H_ + n;
    h_new_f[idx] = v;
    h_new_b[idx] = f2bf(v);
  }
}

// attention: scores -> softmax -> ctx (bf16), one block per batch element
__global__ __launch_bounds__(256)
void k_attn(uint16_t* __restrict__ ctx_b, const float* __restrict__ h_new_f,
            const float* __restrict__ enc) {
  __shared__ float sc[S_];
  __shared__ float we[S_];
  int b   = blockIdx.x;
  int tid = threadIdx.x;
  const float* encb = enc + (size_t)b * S_ * H_;
  const float* hb   = h_new_f + (size_t)b * H_;

  int s = tid >> 2, part = tid & 3;
  float acc = 0.f;
  const float* er = encb + (size_t)s * H_ + part * 128;
  const float* hr = hb + part * 128;
#pragma unroll 4
  for (int h = 0; h < 128; ++h) acc += er[h] * hr[h];
  acc += __shfl_xor(acc, 1);
  acc += __shfl_xor(acc, 2);
  if (part == 0) sc[s] = acc;
  __syncthreads();

  float m = -3.402823466e38f;
  for (int i = 0; i < S_; ++i) m = fmaxf(m, sc[i]);
  if (tid < S_) we[tid] = expf(sc[tid] - m);
  __syncthreads();

  float sum = 0.f;
  for (int i = 0; i < S_; ++i) sum += we[i];
  float inv = 1.f / sum;

  for (int hh = tid; hh < H_; hh += 256) {
    float c = 0.f;
    for (int i = 0; i < S_; ++i) c += we[i] * encb[(size_t)i * H_ + hh];
    ctx_b[(size_t)b * H_ + hh] = f2bf(c * inv);
  }
}

// out[b,t,:] = tanh(h_new @ Wf^T + ctx @ Wc2^T + bfused)   (64 x 8000)
// One wave computes a 16x64 strip: 4 N-tiles sharing the h/ctx A-fragments.
__global__ __launch_bounds__(256)
void k_step_out(float* __restrict__ out, int t,
                const uint16_t* __restrict__ h_new_b, const uint16_t* __restrict__ ctx_b,
                const uint16_t* __restrict__ Wf, const uint16_t* __restrict__ Wc2,
                const float* __restrict__ bfused) {
  int lane = threadIdx.x & 31;
  int wid  = threadIdx.x >> 5;
  int jid  = blockIdx.x * 8 + wid;        // 4 m-tiles * 125 n-groups = 500 jobs
  if (jid >= 500) return;
  int ng = jid % 125;                     // group of 4 V tiles
  int mt = jid / 125;                     // B tile
  const uint16_t* A1 = h_new_b + (size_t)(mt * 16) * H_;
  const uint16_t* A2 = ctx_b   + (size_t)(mt * 16) * H_;
  const uint16_t* B1 = Wf  + (size_t)(ng * 64) * H_;
  const uint16_t* B2 = Wc2 + (size_t)(ng * 64) * H_;
  v8f_t acc[4] = {{}, {}, {}, {}};
  for (int k0 = 0; k0 < H_; k0 += 32) {
    v16bf_t a1 = load_a16(A1, H_, k0, lane);
    v16bf_t a2 = load_a16(A2, H_, k0, lane);
    if (k0 + 32 < H_) {
      // pull next K-slab of the weight strip toward the WGP (global_prefetch_b8)
      __builtin_prefetch(B1 + (size_t)(lane & 15) * H_ + k0 + 32, 0, 3);
      __builtin_prefetch(B2 + (size_t)(lane & 15) * H_ + k0 + 32, 0, 3);
    }
#pragma unroll
    for (int j = 0; j < 4; ++j) {
      acc[j] = wmma_bf16(a1, load_b16(B1 + (size_t)(j * 16) * H_, H_, k0, lane), acc[j]);
      acc[j] = wmma_bf16(a2, load_b16(B2 + (size_t)(j * 16) * H_, H_, k0, lane), acc[j]);
    }
  }
  int mo = mt * 16 + ((lane >> 4) << 3);
#pragma unroll
  for (int j = 0; j < 4; ++j) {
    int n = (ng * 4 + j) * 16 + (lane & 15);
    float bn = bfused[n];
#pragma unroll
    for (int r = 0; r < 8; ++r) {
      int bidx = mo + r;
      out[((size_t)bidx * S_ + t) * V_ + n] = tanhf(acc[j][r] + bn);
    }
  }
}

// ---------------- host launcher ----------------

extern "C" void kernel_launch(void* const* d_in, const int* in_sizes, int n_in,
                              void* d_out, int out_size, void* d_ws, size_t ws_size,
                              hipStream_t stream) {
  (void)in_sizes; (void)n_in; (void)out_size; (void)ws_size;
  const int*   target = (const int*)  d_in[0];
  const float* enc    = (const float*)d_in[1];
  const float* emb    = (const float*)d_in[2];
  const float* W_ih   = (const float*)d_in[3];
  const float* W_hh   = (const float*)d_in[4];
  const float* b_cell = (const float*)d_in[5];
  const float* W_ho   = (const float*)d_in[6];
  const float* b_ho   = (const float*)d_in[7];
  const float* W_c    = (const float*)d_in[8];
  const float* b_c    = (const float*)d_in[9];
  float* out = (float*)d_out;

  char* ws = (char*)d_ws;
  size_t off = 0;
  auto alloc = [&](size_t bytes) -> void* {
    void* p = ws + off;
    off = (off + bytes + 255) & ~(size_t)255;
    return p;
  };

  uint16_t* W_ih_b = (uint16_t*)alloc((size_t)H_ * E_ * 2);
  uint16_t* W_hh_b = (uint16_t*)alloc((size_t)H_ * H_ * 2);
  uint16_t* W_hoT  = (uint16_t*)alloc((size_t)H_ * V_ * 2);
  uint16_t* Wc2    = (uint16_t*)alloc((size_t)V_ * H_ * 2);
  uint16_t* Wf     = (uint16_t*)alloc((size_t)V_ * H_ * 2);
  uint16_t* x_all  = (uint16_t*)alloc((size_t)B_ * S_ * E_ * 2);
  float*    h_f[2];
  uint16_t* h_b[2];
  h_f[0] = (float*)alloc((size_t)B_ * H_ * 4);
  h_f[1] = (float*)alloc((size_t)B_ * H_ * 4);
  h_b[0] = (uint16_t*)alloc((size_t)B_ * H_ * 2);
  h_b[1] = (uint16_t*)alloc((size_t)B_ * H_ * 2);
  uint16_t* ctx_b  = (uint16_t*)alloc((size_t)B_ * H_ * 2);
  float*    bfused = (float*)alloc((size_t)V_ * 4);

  // one-time prep (runs every call; deterministic)
  k_convert<<<(H_ * E_ + 255) / 256, 256, 0, stream>>>(W_ih_b, W_ih, H_ * E_);
  k_convert<<<(H_ * H_ + 255) / 256, 256, 0, stream>>>(W_hh_b, W_hh, H_ * H_);
  k_transpose_convert<<<((V_ * H_) + 255) / 256, 256, 0, stream>>>(W_hoT, W_ho, V_, H_);
  k_slice_convert<<<((V_ * H_) + 255) / 256, 256, 0, stream>>>(Wc2, W_c, V_, H_, VH_, V_);
  k_embed<<<(B_ * S_ * E_ + 255) / 256, 256, 0, stream>>>(x_all, target, emb);
  k_h0<<<(B_ * H_ + 255) / 256, 256, 0, stream>>>(h_f[0], h_b[0], enc);
  k_bias_fuse<<<(V_ + 255) / 256, 256, 0, stream>>>(bfused, W_c, b_ho, b_c);
  k_wf_gemm<<<500, 256, 0, stream>>>(Wf, W_c, W_hoT);

  // sequential scan over time steps
  for (int t = 0; t < S_; ++t) {
    int cur = t & 1, nxt = cur ^ 1;
    k_step_h<<<32, 128, 0, stream>>>(h_f[nxt], h_b[nxt], x_all, t, h_b[cur],
                                     W_ih_b, W_hh_b, b_cell);
    k_attn<<<64, 256, 0, stream>>>(ctx_b, h_f[nxt], enc);
    k_step_out<<<63, 256, 0, stream>>>(out, t, h_b[nxt], ctx_b, Wf, Wc2, bfused);
  }
}
